// LCNN_3289944948918
// MI455X (gfx1250) — matrix operand order repaired and checked by reading, not compiled, not run on hardware
//
#include <hip/hip_runtime.h>
#include <hip/hip_bf16.h>
#include <math.h>

typedef __attribute__((ext_vector_type(16))) _Float16 v16h;
typedef __attribute__((ext_vector_type(8)))  _Float16 v8h;
typedef __attribute__((ext_vector_type(8)))  float    v8f;

#define HDIM 128
#define WDIM 128
#define HW   16384
#define KTOP 150
#define NPAIR 11175   // 150*149/2
#define MSAMP 32

// ---------------------------------------------------------------------------
// Front-end kernels
// ---------------------------------------------------------------------------

// jloc = softmax(hm[0:2],axis=0)[1] = sigmoid(hm1 - hm0)
__global__ void k_jloc(const float* __restrict__ hm, float* __restrict__ jloc) {
    int i = blockIdx.x * blockDim.x + threadIdx.x;
    if (i < HW) jloc[i] = 1.0f / (1.0f + expf(hm[i] - hm[HW + i]));
}

// 3x3 SAME max-pool NMS: jnms = jloc * (jloc == pooled)
__global__ void k_nms(const float* __restrict__ jloc, float* __restrict__ jnms) {
    int i = blockIdx.x * blockDim.x + threadIdx.x;
    if (i >= HW) return;
    int x = i & (WDIM - 1), y = i >> 7;
    float c = jloc[i];
    float m = -INFINITY;
    for (int dy = -1; dy <= 1; ++dy) {
        int yy = y + dy; if (yy < 0 || yy >= HDIM) continue;
        for (int dx = -1; dx <= 1; ++dx) {
            int xx = x + dx; if (xx < 0 || xx >= WDIM) continue;
            float v = jloc[yy * WDIM + xx];
            m = fmaxf(m, v);
        }
    }
    jnms[i] = (c == m) ? c : 0.0f;
}

// Iterative top-K argmax (matches jax top_k ordering: desc value, tie -> low idx).
// Single block, 256 threads, destroys jnms (rebuilt every launch).
__global__ void k_topk(float* jnms, int* __restrict__ topk_idx) {
    __shared__ float sv[256];
    __shared__ int   si[256];
    int tid = threadIdx.x;
    for (int t = 0; t < KTOP; ++t) {
        float bv = -INFINITY; int bi = 0x7fffffff;
        for (int j = 0; j < HW / 256; ++j) {
            int i = j * 256 + tid;
            float v = jnms[i];
            if (v > bv || (v == bv && i < bi)) { bv = v; bi = i; }
        }
        sv[tid] = bv; si[tid] = bi;
        __syncthreads();
        for (int s = 128; s > 0; s >>= 1) {
            if (tid < s) {
                float ov = sv[tid + s]; int oi = si[tid + s];
                if (ov > sv[tid] || (ov == sv[tid] && oi < si[tid])) { sv[tid] = ov; si[tid] = oi; }
            }
            __syncthreads();
        }
        if (tid == 0) { topk_idx[t] = si[0]; jnms[si[0]] = -INFINITY; }
        __syncthreads();
    }
}

// junction coords: jx = idx%W + sigmoid(hm[ch4]), jy = idx/W + sigmoid(hm[ch3])
__global__ void k_junc(const int* __restrict__ topk_idx, const float* __restrict__ hm,
                       float* __restrict__ junc) {
    int j = threadIdx.x;
    if (j < KTOP) {
        int idx = topk_idx[j];
        float offx = 1.0f / (1.0f + expf(-hm[4 * HW + idx]));
        float offy = 1.0f / (1.0f + expf(-hm[3 * HW + idx]));
        junc[2 * j + 0] = (float)(idx & (WDIM - 1)) + offx;
        junc[2 * j + 1] = (float)(idx >> 7) + offy;
    }
}

// all (i<j) pairs in triu row-major order -> line_coords (d_out[0 .. 4*NPAIR))
__global__ void k_pairs(const float* __restrict__ junc, float* __restrict__ out) {
    int p = blockIdx.x * blockDim.x + threadIdx.x;
    if (p >= NPAIR) return;
    int i = 0, rem = p, cnt = KTOP - 1;
    while (rem >= cnt) { rem -= cnt; --cnt; ++i; }
    int j = i + 1 + rem;
    out[4 * p + 0] = junc[2 * i + 0];
    out[4 * p + 1] = junc[2 * i + 1];
    out[4 * p + 2] = junc[2 * j + 0];
    out[4 * p + 3] = junc[2 * j + 1];
}

// ---------------------------------------------------------------------------
// Conversion / transpose kernels
// ---------------------------------------------------------------------------

__global__ void k_f2h(const float* __restrict__ in, _Float16* __restrict__ out, int n) {
    int i = blockIdx.x * blockDim.x + threadIdx.x;
    if (i < n) out[i] = (_Float16)in[i];
}

// features [256][16384] f32 -> featT [16384][256] f16 (LDS-tiled transpose)
__global__ void k_featT(const float* __restrict__ f, _Float16* __restrict__ ft) {
    __shared__ float tile[16][17];
    int p0 = blockIdx.x * 16, c0 = blockIdx.y * 16;
    int tx = threadIdx.x & 15, ty = threadIdx.x >> 4;
    tile[ty][tx] = f[(size_t)(c0 + ty) * HW + p0 + tx];
    __syncthreads();
    ft[(size_t)(p0 + ty) * 256 + c0 + tx] = (_Float16)tile[tx][ty];
}

// ---------------------------------------------------------------------------
// Generic WMMA GEMM:  C[m][n] = sum_k A[m][k]*B[n][k] + bias[n]  (opt ReLU)
// A: f16 row-major (lda), B: f16 row-major (ldb). N tiles of 64.
// K must be a multiple of 64 and >= 64 (true here: 256 and 1024).
// Block = 256 threads (8 waves); block tile = 128(M) x 64(N); wave = 16x64.
// Explicit double-buffered software pipeline, manually unrolled by 2 so there
// is NO register rotation (no v_mov copies): each buffer is consumed by its
// 4 WMMAs and refilled in place; s_wait_loadcnt overlaps ~8 WMMAs of math.
// Out-of-range M rows load via a clamped pointer (garbage only feeds D rows
// that the guarded epilogue never stores), so no per-iteration EXEC masking.
// ---------------------------------------------------------------------------
template <bool RELU, bool OUT_F16>
__global__ __launch_bounds__(256) void gemm_wmma(
    const _Float16* __restrict__ A, int lda,
    const _Float16* __restrict__ B, int ldb,
    const float* __restrict__ bias,
    void* __restrict__ Cout, int ldc,
    int Mdim, int Kdim) {

    const int wave = threadIdx.x >> 5;
    const int lane = threadIdx.x & 31;
    const int half = lane >> 4;        // 0 | 1
    const int l16  = lane & 15;
    const int m0   = blockIdx.x * 128 + wave * 16;
    const int n0b  = blockIdx.y * 64;

    const int mrow = m0 + l16;
    // clamp: invalid rows read row 0; their accumulators are never stored
    const _Float16* arow = A + (size_t)(mrow < Mdim ? mrow : 0) * lda;

    // per-lane fragment base pointers
    // A frag: lanes 0-15 -> K {0..7,16..23}, lanes 16-31 -> K {8..15,24..31}
    const _Float16* apt = arow + half * 8;           // + k0 ; +16 for hi
    // B frag: lane%16 = column, lanes 0-15 -> K {0..7,16..23}, lanes 16-31 -> K {8..15,24..31}
    const _Float16* bpt[4];
#pragma unroll
    for (int j = 0; j < 4; ++j)
        bpt[j] = B + (size_t)(n0b + j * 16 + l16) * ldb + half * 8; // + k0 ; +16 hi

    v8f acc[4] = {{}, {}, {}, {}};

    // fragment load: one A pair + four B pairs -> 10x global_load_b128 clause
    auto load_frag = [&](int k0, v16h& a, v16h (&b)[4]) {
        v8h lo = *(const v8h*)(apt + k0);
        v8h hi = *(const v8h*)(apt + k0 + 16);
        a = __builtin_shufflevector(lo, hi,
            0, 1, 2, 3, 4, 5, 6, 7, 8, 9, 10, 11, 12, 13, 14, 15);
#pragma unroll
        for (int j = 0; j < 4; ++j) {
            v8h blo = *(const v8h*)(bpt[j] + k0);
            v8h bhi = *(const v8h*)(bpt[j] + k0 + 16);
            b[j] = __builtin_shufflevector(blo, bhi,
                0, 1, 2, 3, 4, 5, 6, 7, 8, 9, 10, 11, 12, 13, 14, 15);
        }
    };
    auto mma4 = [&](const v16h& a, const v16h (&b)[4]) {
#pragma unroll
        for (int j = 0; j < 4; ++j)
            acc[j] = __builtin_amdgcn_wmma_f32_16x16x32_f16(
                false, a, false, b[j], (short)0, acc[j], false, false);
    };

    v16h a0, a1, b0[4], b1[4];
    load_frag(0, a0, b0);
    load_frag(32, a1, b1);

    for (int k0 = 64; k0 + 64 <= Kdim; k0 += 64) {
        mma4(a0, b0);
        load_frag(k0, a0, b0);          // WAR: issued right after consume
        mma4(a1, b1);
        load_frag(k0 + 32, a1, b1);
    }
    mma4(a0, b0);                       // drain
    mma4(a1, b1);

    // ---- epilogue: C/D layout -> VGPR r holds M = m0 + r + 8*half ----------
#pragma unroll
    for (int j = 0; j < 4; ++j) {
        int n = n0b + j * 16 + l16;
        float bv = bias[n];
#pragma unroll
        for (int r = 0; r < 8; ++r) {
            int m = m0 + r + half * 8;
            if (m < Mdim) {
                float v = acc[j][r] + bv;
                if (RELU) v = fmaxf(v, 0.0f);
                if (OUT_F16)
                    ((_Float16*)Cout)[(size_t)m * ldc + n] = (_Float16)v;
                else
                    ((float*)Cout)[(size_t)m * ldc + n] = v;
            }
        }
    }
}

// ---------------------------------------------------------------------------
// Bilinear line sampling + maxpool-4 -> lf (f16, [line][c*8+g])
// loiT is [pixel][128ch] f32 so the 128 channel-threads coalesce per corner.
// ---------------------------------------------------------------------------
__global__ __launch_bounds__(128) void k_sample(
    const float* __restrict__ lines,   // d_out line_coords, 4 per line
    const float* __restrict__ loiT,    // [16384][128]
    _Float16* __restrict__ lf) {       // [NPAIR][1024]
    int p = blockIdx.x;
    int c = threadIdx.x;
    float ax = lines[4 * p + 0], ay = lines[4 * p + 1];
    float bx = lines[4 * p + 2], by = lines[4 * p + 3];

    v8h outv;
#pragma unroll
    for (int g = 0; g < 8; ++g) {
        float mv = -INFINITY;
#pragma unroll
        for (int q = 0; q < 4; ++q) {
            int t = g * 4 + q;
            float s  = (float)t * (1.0f / 31.0f);
            float px = ax * s + bx * (1.0f - s) - 0.5f;
            float py = ay * s + by * (1.0f - s) - 0.5f;
            float x0f = fminf(fmaxf(floorf(px), 0.0f), (float)(WDIM - 1));
            float y0f = fminf(fmaxf(floorf(py), 0.0f), (float)(HDIM - 1));
            float x1f = fminf(x0f + 1.0f, (float)(WDIM - 1));
            float y1f = fminf(y0f + 1.0f, (float)(HDIM - 1));
            int x0i = (int)x0f, y0i = (int)y0f, x1i = (int)x1f, y1i = (int)y1f;
            float wx0 = px - x0f, wx1 = x1f - px;
            float wy0 = py - y0f, wy1 = y1f - py;
            float v00 = loiT[(size_t)(y0i * WDIM + x0i) * 128 + c];
            float v10 = loiT[(size_t)(y1i * WDIM + x0i) * 128 + c];
            float v01 = loiT[(size_t)(y0i * WDIM + x1i) * 128 + c];
            float v11 = loiT[(size_t)(y1i * WDIM + x1i) * 128 + c];
            float val = v00 * (wy1 * wx1) + v10 * (wy0 * wx1)
                      + v01 * (wy1 * wx0) + v11 * (wy0 * wx0);
            mv = fmaxf(mv, val);
        }
        outv[g] = (_Float16)mv;
    }
    *(v8h*)(lf + (size_t)p * 1024 + c * 8) = outv;   // 16B coalesced store
}

// ---------------------------------------------------------------------------
// Head: logits = h2 . w3 + b3 ; sigmoid. One wave per line.
// ---------------------------------------------------------------------------
__global__ __launch_bounds__(256) void k_head(
    const _Float16* __restrict__ h2, const float* __restrict__ w3,
    const float* __restrict__ b3, float* __restrict__ scores) {
    int lane = threadIdx.x & 31;
    int wv   = threadIdx.x >> 5;
    int p    = blockIdx.x * 8 + wv;
    if (p >= NPAIR) return;
    const _Float16* row = h2 + (size_t)p * 1024;
    float acc = 0.0f;
#pragma unroll 4
    for (int j = 0; j < 32; ++j) {
        int c = j * 32 + lane;
        acc += (float)row[c] * w3[c];
    }
    for (int off = 16; off > 0; off >>= 1) acc += __shfl_xor(acc, off, 32);
    if (lane == 0) scores[p] = 1.0f / (1.0f + expf(-(acc + b3[0])));
}

// ---------------------------------------------------------------------------
// Host launcher
// ---------------------------------------------------------------------------
extern "C" void kernel_launch(void* const* d_in, const int* in_sizes, int n_in,
                              void* d_out, int out_size, void* d_ws, size_t ws_size,
                              hipStream_t stream) {
    const float* features = (const float*)d_in[0];   // 256*16384
    const float* heatmaps = (const float*)d_in[1];   // 5*16384
    const float* fc1_w    = (const float*)d_in[2];   // 128*256
    const float* fc1_b    = (const float*)d_in[3];   // 128
    const float* w1       = (const float*)d_in[4];   // 1024*1032
    const float* b1       = (const float*)d_in[5];   // 1024
    const float* w2       = (const float*)d_in[6];   // 1024*1024
    const float* b2       = (const float*)d_in[7];   // 1024
    const float* w3       = (const float*)d_in[8];   // 1024
    const float* b3       = (const float*)d_in[9];   // 1
    float* out = (float*)d_out;                      // 4*NPAIR coords | NPAIR scores

    // workspace carve-up (256B aligned slabs)
    char*  base = (char*)d_ws;
    size_t off  = 0;
    auto alloc = [&](size_t bytes) -> void* {
        void* p = base + off;
        off = (off + bytes + 255) & ~(size_t)255;
        return p;
    };
    int*       topk_idx = (int*)      alloc(KTOP * sizeof(int));
    float*     jloc     = (float*)    alloc(HW * sizeof(float));
    float*     jnms     = (float*)    alloc(HW * sizeof(float));
    float*     junc     = (float*)    alloc(2 * KTOP * sizeof(float));
    _Float16*  fc1wh    = (_Float16*) alloc((size_t)128 * 256 * 2);
    _Float16*  w1h      = (_Float16*) alloc((size_t)1024 * 1032 * 2);
    _Float16*  w2h      = (_Float16*) alloc((size_t)1024 * 1024 * 2);
    _Float16*  featT    = (_Float16*) alloc((size_t)HW * 256 * 2);
    float*     loiT     = (float*)    alloc((size_t)HW * 128 * 4);
    _Float16*  lf       = (_Float16*) alloc((size_t)NPAIR * 1024 * 2);  // reused as h2
    _Float16*  h1       = (_Float16*) alloc((size_t)NPAIR * 1024 * 2);
    _Float16*  h2       = lf;  // MLP2 writes over lf (no longer needed)

    // 1) front end
    k_jloc<<<HW / 256, 256, 0, stream>>>(heatmaps, jloc);
    k_nms <<<HW / 256, 256, 0, stream>>>(jloc, jnms);
    k_topk<<<1, 256, 0, stream>>>(jnms, topk_idx);
    k_junc<<<1, 256, 0, stream>>>(topk_idx, heatmaps, junc);
    k_pairs<<<(NPAIR + 255) / 256, 256, 0, stream>>>(junc, out);

    // 2) fp16 conversions + feature transpose
    k_f2h<<<(128 * 256) / 256, 256, 0, stream>>>(fc1_w, fc1wh, 128 * 256);
    k_f2h<<<(1024 * 1032) / 256, 256, 0, stream>>>(w1, w1h, 1024 * 1032);
    k_f2h<<<(1024 * 1024) / 256, 256, 0, stream>>>(w2, w2h, 1024 * 1024);
    k_featT<<<dim3(HW / 16, 256 / 16), 256, 0, stream>>>(features, featT);

    // 3) loi GEMM: [16384 px][256 c] x [128 o][256 c]^T -> loiT [px][o] (f32)
    gemm_wmma<false, false><<<dim3(HW / 128, 128 / 64), 256, 0, stream>>>(
        featT, 256, fc1wh, 256, fc1_b, loiT, 128, HW, 256);

    // 4) line sampling -> lf (f16)
    k_sample<<<NPAIR, 128, 0, stream>>>(out, loiT, lf);

    // 5) MLP (zero-pad cols of lf are zero => K=1024; w1 row stride stays 1032)
    gemm_wmma<true, true><<<dim3((NPAIR + 127) / 128, 1024 / 64), 256, 0, stream>>>(
        lf, 1024, w1h, 1032, b1, h1, 1024, NPAIR, 1024);
    gemm_wmma<true, true><<<dim3((NPAIR + 127) / 128, 1024 / 64), 256, 0, stream>>>(
        h1, 1024, w2h, 1024, b2, h2, 1024, NPAIR, 1024);

    // 6) head -> scores
    k_head<<<(NPAIR + 7) / 8, 256, 0, stream>>>(h2, w3, b3, out + 4 * NPAIR);
}
